// VRWKV_Multi_SpatialMix_12249246728666
// MI455X (gfx1250) — matrix operand order repaired and checked by reading, not compiled, not run on hardware
//
#include <hip/hip_runtime.h>
#include <hip/hip_bf16.h>

// ---------------------------------------------------------------------------
// VRWKV SpatialMix for MI455X (gfx1250, wave32).
//   B=8, T=2304 (=48x48), C=192, HEAD=4 -> C_BIG=768.
// Pipeline:
//   1. convert_f32_to_f16: Wk|Wv|Wr -> Wh [2304x192] f16, Wout -> Wout_h f16
//   2. omnishift: depthwise 1/3/5 conv mix, writes xs as f16 [B*T, 192]
//   3. gemm_kvr: v_wmma_f32_16x16x32_f16, 16x64 tile/wave, software-pipelined
//   4. wkv_scan: serial T recurrence per (b,c), fp32 + v_exp_f32, branch-free
//      prefetch steady-state; writes ys = y*sr as f16 [B*T, 768]
//   5. gemm_out: v_wmma_f32_16x16x32_f16, 16x64 tile/wave, software-pipelined
// ---------------------------------------------------------------------------

#define BB   8
#define TT   2304
#define CE   192
#define CBIG 768
#define HH   48
#define WW2  48

typedef __attribute__((ext_vector_type(16))) _Float16 v16h;
typedef __attribute__((ext_vector_type(8)))  _Float16 v8h;
typedef __attribute__((ext_vector_type(8)))  float    v8f;

#define WMMA_F16(a, b, c) \
    __builtin_amdgcn_wmma_f32_16x16x32_f16(false, (a), false, (b), (short)0, (c), false, false)

// ---------------------------------------------------------------- convert ---
__global__ void convert_f32_to_f16(const float* __restrict__ src,
                                   _Float16* __restrict__ dst, int n) {
    int i = blockIdx.x * blockDim.x + threadIdx.x;
    if (i < n) dst[i] = (_Float16)src[i];
}

// -------------------------------------------------------------- omnishift ---
// block: 192 threads (c), grid: (T, B). All loads coalesced across c.
__global__ __launch_bounds__(CE) void omnishift(
    const float* __restrict__ x, const float* __restrict__ alpha,
    const float* __restrict__ w1, const float* __restrict__ w3,
    const float* __restrict__ w5, _Float16* __restrict__ xs_h) {
    const int c = threadIdx.x;
    const int t = blockIdx.x;
    const int b = blockIdx.y;
    const int h0 = t / WW2, w0 = t % WW2;
    const float a0 = alpha[0], a1 = alpha[1], a2 = alpha[2], a3 = alpha[3];
    const float* xb = x + (size_t)b * TT * CE + c;

    float acc = (a0 + a1 * w1[c]) * xb[(size_t)t * CE];

    // 3x3 (cross-correlation, zero pad 1)
    #pragma unroll
    for (int dh = -1; dh <= 1; ++dh)
        #pragma unroll
        for (int dw = -1; dw <= 1; ++dw) {
            int hh = h0 + dh, ww = w0 + dw;
            if (hh >= 0 && hh < HH && ww >= 0 && ww < WW2) {
                float v = xb[(size_t)(hh * WW2 + ww) * CE];
                acc += a2 * w3[c * 9 + (dh + 1) * 3 + (dw + 1)] * v;
            }
        }
    // 5x5 (zero pad 2)
    #pragma unroll
    for (int dh = -2; dh <= 2; ++dh)
        #pragma unroll
        for (int dw = -2; dw <= 2; ++dw) {
            int hh = h0 + dh, ww = w0 + dw;
            if (hh >= 0 && hh < HH && ww >= 0 && ww < WW2) {
                float v = xb[(size_t)(hh * WW2 + ww) * CE];
                acc += a3 * w5[c * 25 + (dh + 2) * 5 + (dw + 2)] * v;
            }
        }
    xs_h[((size_t)b * TT + t) * CE + c] = (_Float16)acc;
}

// ----------------------------------------------------------- WMMA helpers ---
// A fragment (16x32 f16, row-major source, leading dim lda in halves):
// lane L: row m0 + (L&15); halves 0..15 = K kk+16*(L>>4)..+15 (matches B)
__device__ inline v16h load_a_frag(const _Float16* __restrict__ A, int lda,
                                   int m0, int kk, int lane) {
    const _Float16* p = A + (size_t)(m0 + (lane & 15)) * lda + kk + 16 * (lane >> 4);
    return *(const v16h*)p;
}
// B fragment (32x16 f16): lane L holds column n0+(L&15), 16 contiguous K
// starting at kk+16*(L>>4). W is [N,K] row-major -> contiguous load.
__device__ inline v16h load_b_frag(const _Float16* __restrict__ W, int ldw,
                                   int n0, int kk, int lane) {
    return *(const v16h*)(W + (size_t)(n0 + (lane & 15)) * ldw + kk + 16 * (lane >> 4));
}

// --------------------------------------------------------------- gemm_kvr ---
// C = xs[18432x192] * Wh^T, Wh = [Wk;Wv;Wr] (2304x192).
// One 16x64 tile/wave, software-pipelined: next K-step fragments load into
// fresh registers before the current WMMA quad issues (guards const-fold
// under full unroll). waves_per_eu(1,8) lifts the ~64-VGPR occupancy cap so
// double-buffering (~128 VGPRs) is possible.
// grid (1152, 9), block 128 (4 waves, each owns one 64-wide N-supertile).
__global__ __launch_bounds__(128)
__attribute__((amdgpu_waves_per_eu(1, 8)))
void gemm_kvr(const _Float16* __restrict__ A, const _Float16* __restrict__ W,
              float* __restrict__ kbuf, float* __restrict__ vbuf,
              _Float16* __restrict__ srbuf) {
    const int lane = threadIdx.x & 31;
    const int wid  = threadIdx.x >> 5;
    const int m0 = blockIdx.x * 16;
    const int n0 = (blockIdx.y * 4 + wid) * 64;   // 0..2240, 64 | 768

    v8f acc0 = {}, acc1 = {}, acc2 = {}, acc3 = {};
    v16h a  = load_a_frag(A, CE, m0, 0, lane);
    v16h b0 = load_b_frag(W, CE, n0,      0, lane);
    v16h b1 = load_b_frag(W, CE, n0 + 16, 0, lane);
    v16h b2 = load_b_frag(W, CE, n0 + 32, 0, lane);
    v16h b3 = load_b_frag(W, CE, n0 + 48, 0, lane);
    #pragma unroll
    for (int kk = 0; kk < CE; kk += 32) {
        v16h an, bn0, bn1, bn2, bn3;
        if (kk + 32 < CE) {                       // const-folded (full unroll)
            an  = load_a_frag(A, CE, m0, kk + 32, lane);
            bn0 = load_b_frag(W, CE, n0,      kk + 32, lane);
            bn1 = load_b_frag(W, CE, n0 + 16, kk + 32, lane);
            bn2 = load_b_frag(W, CE, n0 + 32, kk + 32, lane);
            bn3 = load_b_frag(W, CE, n0 + 48, kk + 32, lane);
        }
        acc0 = WMMA_F16(a, b0, acc0);
        acc1 = WMMA_F16(a, b1, acc1);
        acc2 = WMMA_F16(a, b2, acc2);
        acc3 = WMMA_F16(a, b3, acc3);
        if (kk + 32 < CE) { a = an; b0 = bn0; b1 = bn1; b2 = bn2; b3 = bn3; }
    }

    const int rbase = m0 + 8 * (lane >> 4);
    const int ncol  = (n0 % CBIG) + (lane & 15);
    const int sel   = n0 / CBIG;                  // uniform per wave
    v8f accs[4] = {acc0, acc1, acc2, acc3};
    if (sel == 0) {
        #pragma unroll
        for (int j = 0; j < 4; ++j)
            #pragma unroll
            for (int r = 0; r < 8; ++r)
                kbuf[(size_t)(rbase + r) * CBIG + ncol + j * 16] = accs[j][r];
    } else if (sel == 1) {
        #pragma unroll
        for (int j = 0; j < 4; ++j)
            #pragma unroll
            for (int r = 0; r < 8; ++r)
                vbuf[(size_t)(rbase + r) * CBIG + ncol + j * 16] = accs[j][r];
    } else {
        #pragma unroll
        for (int j = 0; j < 4; ++j)
            #pragma unroll
            for (int r = 0; r < 8; ++r) {
                float s = 1.0f / (1.0f + __expf(-accs[j][r]));
                srbuf[(size_t)(rbase + r) * CBIG + ncol + j * 16] = (_Float16)s;
            }
    }
}

// --------------------------------------------------------------- wkv_scan ---
// One thread per (b, c) channel; consecutive threads -> consecutive c
// (coalesced 128B/wave per array per step). fp32 log-sum-exp state.
__global__ __launch_bounds__(256) void wkv_scan(
    const float* __restrict__ kbuf, const float* __restrict__ vbuf,
    const _Float16* __restrict__ srbuf, const float* __restrict__ sdecay,
    const float* __restrict__ sfirst, _Float16* __restrict__ ys) {
    const int tid = blockIdx.x * blockDim.x + threadIdx.x;   // 0..6143
    const int b = tid / CBIG;
    const int c = tid % CBIG;
    const size_t base = (size_t)b * TT * CBIG + c;
    const float* kp = kbuf + base;
    const float* vp = vbuf + base;
    const _Float16* sp = srbuf + base;
    _Float16* yp = ys + base;

    const float u = sfirst[c] * (1.0f / TT);
    const float w = -__expf(sdecay[c] * (1.0f / TT));

    float aa = 0.0f, bb = 0.0f, pp = -1e38f;
    #define WKV_STEP(t)                                              \
        {                                                            \
            const size_t o = (size_t)(t) * CBIG;                     \
            float kt = kp[o], vt = vp[o];                            \
            float wwk = u + kt;                                      \
            float p   = fmaxf(pp, wwk);                              \
            float e1  = __expf(pp - p);                              \
            float e2  = __expf(wwk - p);                             \
            float y   = (e1 * aa + e2 * vt) / (e1 * bb + e2);        \
            float ww2 = pp + w;                                      \
            float p2  = fmaxf(ww2, kt);                              \
            float e1b = __expf(ww2 - p2);                            \
            float e2b = __expf(kt - p2);                             \
            aa = e1b * aa + e2b * vt;                                \
            bb = e1b * bb + e2b;                                     \
            pp = p2;                                                 \
            yp[o] = (_Float16)(y * (float)sp[o]);                    \
        }
    int t = 0;
    for (; t < TT - 16; ++t) {                   // branch-free steady state
        __builtin_prefetch(kp + (size_t)(t + 16) * CBIG, 0, 1);
        __builtin_prefetch(vp + (size_t)(t + 16) * CBIG, 0, 1);
        WKV_STEP(t);
    }
    for (; t < TT; ++t) WKV_STEP(t);             // tail, no prefetch
    #undef WKV_STEP
}

// --------------------------------------------------------------- gemm_out ---
// out = ys[18432x768] * Wout^T (Wout_h [192x768]).
// One 16x64 tile/wave; 3 waves cover N=192.  Prologue + 23-step steady loop
// (no conditionals in body) + epilogue WMMAs.  grid (1152,1), block 96.
__global__ __launch_bounds__(96)
__attribute__((amdgpu_waves_per_eu(1, 8)))
void gemm_out(const _Float16* __restrict__ A, const _Float16* __restrict__ W,
              float* __restrict__ out) {
    const int lane = threadIdx.x & 31;
    const int wid  = threadIdx.x >> 5;            // 0..2
    const int m0 = blockIdx.x * 16;
    const int n0 = wid * 64;                      // 0, 64, 128

    v8f acc0 = {}, acc1 = {}, acc2 = {}, acc3 = {};
    v16h a  = load_a_frag(A, CBIG, m0, 0, lane);
    v16h b0 = load_b_frag(W, CBIG, n0,      0, lane);
    v16h b1 = load_b_frag(W, CBIG, n0 + 16, 0, lane);
    v16h b2 = load_b_frag(W, CBIG, n0 + 32, 0, lane);
    v16h b3 = load_b_frag(W, CBIG, n0 + 48, 0, lane);
    #pragma unroll 2
    for (int kk = 32; kk < CBIG; kk += 32) {      // steady state: load next,
        v16h an  = load_a_frag(A, CBIG, m0, kk, lane);   // compute current
        v16h bn0 = load_b_frag(W, CBIG, n0,      kk, lane);
        v16h bn1 = load_b_frag(W, CBIG, n0 + 16, kk, lane);
        v16h bn2 = load_b_frag(W, CBIG, n0 + 32, kk, lane);
        v16h bn3 = load_b_frag(W, CBIG, n0 + 48, kk, lane);
        acc0 = WMMA_F16(a, b0, acc0);
        acc1 = WMMA_F16(a, b1, acc1);
        acc2 = WMMA_F16(a, b2, acc2);
        acc3 = WMMA_F16(a, b3, acc3);
        a = an; b0 = bn0; b1 = bn1; b2 = bn2; b3 = bn3;
    }
    acc0 = WMMA_F16(a, b0, acc0);                 // epilogue (last K-step)
    acc1 = WMMA_F16(a, b1, acc1);
    acc2 = WMMA_F16(a, b2, acc2);
    acc3 = WMMA_F16(a, b3, acc3);

    const int rbase = m0 + 8 * (lane >> 4);
    const int ncol  = n0 + (lane & 15);
    v8f accs[4] = {acc0, acc1, acc2, acc3};
    #pragma unroll
    for (int j = 0; j < 4; ++j)
        #pragma unroll
        for (int r = 0; r < 8; ++r)
            out[(size_t)(rbase + r) * CE + ncol + j * 16] = accs[j][r];
}

// ----------------------------------------------------------------- launch ---
extern "C" void kernel_launch(void* const* d_in, const int* in_sizes, int n_in,
                              void* d_out, int out_size, void* d_ws, size_t ws_size,
                              hipStream_t stream) {
    const float* x      = (const float*)d_in[0];
    const float* alpha  = (const float*)d_in[1];
    const float* w1     = (const float*)d_in[2];
    const float* w3     = (const float*)d_in[3];
    const float* w5     = (const float*)d_in[4];
    const float* Wk     = (const float*)d_in[5];
    const float* Wv     = (const float*)d_in[6];
    const float* Wr     = (const float*)d_in[7];
    const float* Wout   = (const float*)d_in[8];
    const float* sdecay = (const float*)d_in[9];
    const float* sfirst = (const float*)d_in[10];
    float* out = (float*)d_out;

    // workspace carve-up (256B aligned)
    char* ws = (char*)d_ws;
    size_t off = 0;
    auto carve = [&](size_t bytes) {
        char* p = ws + off;
        off = (off + bytes + 255) & ~(size_t)255;
        return p;
    };
    const size_t NW   = (size_t)CBIG * CE;        // 147456
    const size_t MROW = (size_t)BB * TT;          // 18432
    _Float16* xs_h   = (_Float16*)carve(MROW * CE * 2);
    _Float16* Wh     = (_Float16*)carve(3 * NW * 2);
    _Float16* Wout_h = (_Float16*)carve(NW * 2);
    float*    kbuf   = (float*)carve(MROW * CBIG * 4);
    float*    vbuf   = (float*)carve(MROW * CBIG * 4);
    _Float16* srbuf  = (_Float16*)carve(MROW * CBIG * 2);
    _Float16* ys     = (_Float16*)carve(MROW * CBIG * 2);

    // 1. weight conversion (f32 -> f16)
    {
        int n = (int)NW, blk = 256, g = (n + blk - 1) / blk;
        convert_f32_to_f16<<<g, blk, 0, stream>>>(Wk,   Wh,          n);
        convert_f32_to_f16<<<g, blk, 0, stream>>>(Wv,   Wh + NW,     n);
        convert_f32_to_f16<<<g, blk, 0, stream>>>(Wr,   Wh + 2 * NW, n);
        convert_f32_to_f16<<<g, blk, 0, stream>>>(Wout, Wout_h,      n);
    }
    // 2. OmniShift + f16 pack
    omnishift<<<dim3(TT, BB), CE, 0, stream>>>(x, alpha, w1, w3, w5, xs_h);
    // 3. K/V/sigmoid(R) GEMM  (M=18432, N=2304, K=192), 16x64 tile/wave
    gemm_kvr<<<dim3(MROW / 16, (3 * CBIG / 64) / 4), 128, 0, stream>>>(
        xs_h, Wh, kbuf, vbuf, srbuf);
    // 4. WKV recurrence (serial T, parallel over 6144 channels)
    wkv_scan<<<(BB * CBIG) / 256, 256, 0, stream>>>(
        kbuf, vbuf, srbuf, sdecay, sfirst, ys);
    // 5. output GEMM  (M=18432, N=192, K=768), 16x64 tile/wave
    gemm_out<<<dim3(MROW / 16, 1), 96, 0, stream>>>(ys, Wout_h, out);
    (void)in_sizes; (void)n_in; (void)out_size; (void)ws_size;
}
